// NeRF_67826123539074
// MI455X (gfx1250) — matrix operand order repaired and checked by reading, not compiled, not run on hardware
//
#include <hip/hip_runtime.h>
#include <hip/hip_bf16.h>

typedef __attribute__((ext_vector_type(16))) _Float16 v16h;
typedef __attribute__((ext_vector_type(8)))  _Float16 v8h;
typedef __attribute__((ext_vector_type(8)))  float    v8f;
typedef __attribute__((ext_vector_type(4)))  int      v4i;

// ---- geometry ----
#define XYZ      63
#define INF      90      // x feature count (63 + 27 + 0)
#define HID      256
#define WAVES    8       // waves per block
#define ROWS_PB  (16 * WAVES)   // 128 batch rows per block

// padded LDS strides (halfs); all multiples of 8 halfs (16B) for b128 alignment
#define WPAD 40          // wslab row stride  (32 K + pad)
#define XPAD 72          // xbuf  row stride  (64 K + pad)
#define HPAD 264         // hbuf  row stride  (256 K + pad)
#define SLAB (256 * WPAD)

// f16 weight offsets inside d_ws (in halfs); layout Wt[n][Kpad] row-major
#define OFF_W0    0          // [256][64]
#define OFF_W1    16384      // [256][256]
#define OFF_W2    81920
#define OFF_W3    147456
#define OFF_W4    212992
#define OFF_W5    278528     // [256][320]  (k=63 zero row; 64..319 = h part)
#define OFF_W6    360448
#define OFF_W7    425984
#define OFF_WOUT  491520     // [16][256]   (n>=4 zero)

// ---- CDNA5 async global->LDS staging (GLOBAL_LOAD_ASYNC_TO_LDS_B128, ASYNCcnt)
#if defined(__HIP_DEVICE_COMPILE__) && defined(__gfx1250__)
#if __has_builtin(__builtin_amdgcn_global_load_async_to_lds_b128) && \
    __has_builtin(__builtin_amdgcn_s_wait_asynccnt)
#define USE_ASYNC 1
#endif
#endif

#if defined(USE_ASYNC)
typedef __attribute__((address_space(1))) v4i as1_v4i;   // global int4*
typedef __attribute__((address_space(3))) v4i as3_v4i;   // LDS int4*
#define ASYNC_WAIT() __builtin_amdgcn_s_wait_asynccnt(0)
#else
#define ASYNC_WAIT() ((void)0)
#endif

// ---- scheduler pinning: interleave DS-reads with WMMAs (prefetch distance 2)
#if defined(__HIP_DEVICE_COMPILE__) && __has_builtin(__builtin_amdgcn_sched_group_barrier)
#define HAVE_SGB 1
#define SGB(mask, n) __builtin_amdgcn_sched_group_barrier(mask, n, 0)
#else
#define SGB(mask, n) ((void)0)
#endif

// ---------------------------------------------------------------------------
// Pre-pass: convert f32 weights -> f16, transposed to Wt[n][k], K/N padded.
// zpad=1 inserts a zero row at k==63 and shifts k>63 down by 1 (skip layer).
// ---------------------------------------------------------------------------
__global__ void cvt_weight(const float* __restrict__ src, _Float16* __restrict__ dst,
                           int srcRows, int srcCols, int Kdim, int total, int zpad)
{
    int i = blockIdx.x * blockDim.x + threadIdx.x;
    if (i >= total) return;
    int n = i / Kdim;
    int k = i - n * Kdim;
    int sr = k;
    bool zero = false;
    if (zpad) {
        if (k == 63)      zero = true;
        else if (k > 63)  sr = k - 1;
    }
    float v = 0.0f;
    if (!zero && sr < srcRows && n < srcCols)
        v = src[(size_t)sr * srcCols + n];
    dst[i] = (_Float16)v;
}

// stage one 32-K slab (row n = tid) of pre-transposed weights into an LDS buffer
__device__ __forceinline__ void stage_slab(const _Float16* __restrict__ wt, int Kdim,
                                           int ks, _Float16* __restrict__ dstbase)
{
    const _Float16* srcp = wt + (size_t)threadIdx.x * Kdim + ks * 32;
    _Float16* drow = dstbase + threadIdx.x * WPAD;
#if defined(USE_ASYNC)
    __builtin_amdgcn_global_load_async_to_lds_b128((as1_v4i*)srcp, (as3_v4i*)drow,  0, 0);
    __builtin_amdgcn_global_load_async_to_lds_b128((as1_v4i*)srcp, (as3_v4i*)drow, 16, 0);
    __builtin_amdgcn_global_load_async_to_lds_b128((as1_v4i*)srcp, (as3_v4i*)drow, 32, 0);
    __builtin_amdgcn_global_load_async_to_lds_b128((as1_v4i*)srcp, (as3_v4i*)drow, 48, 0);
#else
    *(v8h*)(drow +  0) = *(const v8h*)(srcp +  0);
    *(v8h*)(drow +  8) = *(const v8h*)(srcp +  8);
    *(v8h*)(drow + 16) = *(const v8h*)(srcp + 16);
    *(v8h*)(drow + 24) = *(const v8h*)(srcp + 24);
#endif
}

// ---------------------------------------------------------------------------
// Fused NeRF MLP: 8 hidden layers + skip(concat @ layer 5 input) + out layer.
// One wave32 owns 16 batch rows; 8 waves/block share double-buffered weight
// slabs staged LDS-async; B operands double-buffered in VGPRs.
// ---------------------------------------------------------------------------
__global__ void __launch_bounds__(256)
nerf_mlp(const float* __restrict__ x,
         const _Float16* __restrict__ wts,
         const float* __restrict__ b0, const float* __restrict__ b1,
         const float* __restrict__ b2, const float* __restrict__ b3,
         const float* __restrict__ b4, const float* __restrict__ b5,
         const float* __restrict__ b6, const float* __restrict__ b7,
         const float* __restrict__ bout,
         float* __restrict__ out)
{
    __shared__ alignas(16) _Float16 s_wslab[2 * SLAB];            // 40 KB (double buffer)
    __shared__ alignas(16) _Float16 s_xbuf [WAVES * 16 * XPAD];   // 18 KB
    __shared__ alignas(16) _Float16 s_hbuf [WAVES * 16 * HPAD];   // 66 KB

    const int lane = threadIdx.x & 31;
    const int wave = threadIdx.x >> 5;
    const int rowBase = blockIdx.x * ROWS_PB + wave * 16;
    const int lmod = lane & 15;         // row-within-tile / col-within-tile
    const int lhi  = (lane >> 4) & 1;   // half-wave select

    _Float16* xw = s_xbuf + wave * 16 * XPAD;
    _Float16* hw = s_hbuf + wave * 16 * HPAD;

    // ---- load input points (first 63 of 90 features), pad K 63..71 with 0 ----
    {
        const float* xg = x + (size_t)rowBase * INF;
        for (int c = lane; c < XPAD; c += 32) {
            #pragma unroll
            for (int m = 0; m < 16; ++m) {
                float v = (c < XYZ) ? xg[m * INF + c] : 0.0f;
                xw[m * XPAD + c] = (_Float16)v;
            }
        }
    }

    union AB { v16h v; v8h h[2]; };

    // ---- 8 hidden layers (every layer has an EVEN K-step count: 2/8/10/8) ----
    for (int layer = 0; layer < 8; ++layer) {
        const int nk   = (layer == 0) ? 2 : (layer == 5) ? 10 : 8;
        const int Kdim = nk * 32;
        const int woff = (layer == 0) ? OFF_W0 :
                         (layer == 5) ? OFF_W5 :
                         (layer == 6) ? OFF_W6 :
                         (layer == 7) ? OFF_W7 : OFF_W1 + (layer - 1) * 65536;
        const float* bl = (layer == 0) ? b0 : (layer == 1) ? b1 : (layer == 2) ? b2 :
                          (layer == 3) ? b3 : (layer == 4) ? b4 : (layer == 5) ? b5 :
                          (layer == 6) ? b6 : b7;
        const _Float16* wt = wts + woff;

        v8f acc[16];
        #pragma unroll
        for (int nt = 0; nt < 16; ++nt) acc[nt] = (v8f){0,0,0,0,0,0,0,0};

        // prologue: stage slab 0 into buffer 0
        stage_slab(wt, Kdim, 0, s_wslab);
        ASYNC_WAIT();
        __syncthreads();

        for (int ks = 0; ks < nk; ++ks) {
            _Float16* wsl = s_wslab + (ks & 1) * SLAB;
            if (ks + 1 < nk)                       // async-prefetch next slab
                stage_slab(wt, Kdim, ks + 1, s_wslab + ((ks + 1) & 1) * SLAB);

            // A operand: 16x32 f16 tile in ISA register layout
            const _Float16* abase;
            int kl, astr;
            if (layer == 0 || (layer == 5 && ks < 2)) { abase = xw; kl = ks * 32;       astr = XPAD; }
            else if (layer == 5)                      { abase = hw; kl = (ks - 2) * 32; astr = HPAD; }
            else                                      { abase = hw; kl = ks * 32;       astr = HPAD; }
            const _Float16* ap = abase + lmod * astr + kl + (lhi << 3);
            AB A;
            A.h[0] = *(const v8h*)(ap);        // K j=0..7  -> k0 + 8*lhi + j
            A.h[1] = *(const v8h*)(ap + 16);   // K j=8..15 -> k0 + 16 + 8*lhi + (j-8)

            // B operands double-buffered in VGPRs: prefetch nt+1 under WMMA nt
            const _Float16* bbase = wsl + lmod * WPAD + (lhi << 3);
            AB Bb[2];
            Bb[0].h[0] = *(const v8h*)(bbase);
            Bb[0].h[1] = *(const v8h*)(bbase + 16);
            #pragma unroll
            for (int nt = 0; nt < 16; ++nt) {
                if (nt < 15) {
                    const _Float16* bp = bbase + (nt + 1) * (16 * WPAD);
                    Bb[(nt + 1) & 1].h[0] = *(const v8h*)(bp);
                    Bb[(nt + 1) & 1].h[1] = *(const v8h*)(bp + 16);
                }
                acc[nt] = __builtin_amdgcn_wmma_f32_16x16x32_f16(
                    false, A.v, false, Bb[nt & 1].v, (short)0, acc[nt], false, false);
            }

#if defined(HAVE_SGB)
            // Pin schedule: A(2)+B0(2)+B1(2) DS reads up front, then WMMA n
            // interleaved with the DS-read pair of B(n+2)  ->  dscnt waits of 2,
            // LDS latency hidden under the matrix pipe.
            SGB(0x100, 6);
            #pragma unroll
            for (int q = 0; q < 14; ++q) { SGB(0x8, 1); SGB(0x100, 2); }
            SGB(0x8, 1);
            SGB(0x8, 1);
#endif

            if (ks + 1 < nk) {                 // next slab must have landed, all waves
                ASYNC_WAIT();
                __syncthreads();
            }
        }

        // epilogue: bias + ReLU, f32 -> f16, into per-wave h buffer
        #pragma unroll
        for (int nt = 0; nt < 16; ++nt) {
            float bv = bl[nt * 16 + lmod];
            #pragma unroll
            for (int r = 0; r < 8; ++r) {
                float v = acc[nt][r] + bv;
                v = v > 0.0f ? v : 0.0f;
                hw[(r + (lhi << 3)) * HPAD + nt * 16 + lmod] = (_Float16)v;
            }
        }
    }

    // ---- output layer: 256 -> 4 (padded to one 16-wide N tile) ----
    {
        const _Float16* wt = wts + OFF_WOUT;
        v8f acc = (v8f){0,0,0,0,0,0,0,0};

        if (threadIdx.x < 16) stage_slab(wt, 256, 0, s_wslab);
        ASYNC_WAIT();
        __syncthreads();

        for (int ks = 0; ks < 8; ++ks) {
            _Float16* wsl = s_wslab + (ks & 1) * SLAB;
            if (ks + 1 < 8 && threadIdx.x < 16)
                stage_slab(wt, 256, ks + 1, s_wslab + ((ks + 1) & 1) * SLAB);

            const _Float16* ap = hw + lmod * HPAD + ks * 32 + (lhi << 3);
            AB A;
            A.h[0] = *(const v8h*)(ap);
            A.h[1] = *(const v8h*)(ap + 16);
            const _Float16* bp = wsl + lmod * WPAD + (lhi << 3);
            AB B;
            B.h[0] = *(const v8h*)(bp);
            B.h[1] = *(const v8h*)(bp + 16);
            acc = __builtin_amdgcn_wmma_f32_16x16x32_f16(
                false, A.v, false, B.v, (short)0, acc, false, false);

            if (ks + 1 < 8) {
                ASYNC_WAIT();
                __syncthreads();
            }
        }

        const int c = lmod;            // output column this lane holds
        if (c < 4) {
            const float bv = bout[c];
            #pragma unroll
            for (int r = 0; r < 8; ++r) {
                int row = rowBase + r + (lhi << 3);
                out[(size_t)row * 4 + c] = acc[r] + bv;
            }
        }
    }
}

// ---------------------------------------------------------------------------
extern "C" void kernel_launch(void* const* d_in, const int* in_sizes, int n_in,
                              void* d_out, int out_size, void* d_ws, size_t ws_size,
                              hipStream_t stream)
{
    const float* x = (const float*)d_in[0];
    const float* W[8]; const float* b[8];
    for (int i = 0; i < 8; ++i) { W[i] = (const float*)d_in[1 + 2*i]; b[i] = (const float*)d_in[2 + 2*i]; }
    const float* Wout = (const float*)d_in[17];
    const float* bout = (const float*)d_in[18];
    float* outp = (float*)d_out;
    _Float16* ws = (_Float16*)d_ws;

    // ---- pre-pass: f32 -> f16 transposed, K-padded weights into d_ws ----
    cvt_weight<<<(256*64 + 255)/256, 256, 0, stream>>>(W[0], ws + OFF_W0, 63, 256, 64, 256*64, 0);
    cvt_weight<<<(65536 + 255)/256, 256, 0, stream>>>(W[1], ws + OFF_W1, 256, 256, 256, 65536, 0);
    cvt_weight<<<(65536 + 255)/256, 256, 0, stream>>>(W[2], ws + OFF_W2, 256, 256, 256, 65536, 0);
    cvt_weight<<<(65536 + 255)/256, 256, 0, stream>>>(W[3], ws + OFF_W3, 256, 256, 256, 65536, 0);
    cvt_weight<<<(65536 + 255)/256, 256, 0, stream>>>(W[4], ws + OFF_W4, 256, 256, 256, 65536, 0);
    cvt_weight<<<(256*320 + 255)/256, 256, 0, stream>>>(W[5], ws + OFF_W5, 319, 256, 320, 256*320, 1);
    cvt_weight<<<(65536 + 255)/256, 256, 0, stream>>>(W[6], ws + OFF_W6, 256, 256, 256, 65536, 0);
    cvt_weight<<<(65536 + 255)/256, 256, 0, stream>>>(W[7], ws + OFF_W7, 256, 256, 256, 65536, 0);
    cvt_weight<<<(16*256 + 255)/256, 256, 0, stream>>>(Wout, ws + OFF_WOUT, 256, 4, 256, 16*256, 0);

    // ---- fused MLP ----
    const int nRows  = in_sizes[0] / INF;       // 262144
    const int blocks = nRows / ROWS_PB;         // 2048
    nerf_mlp<<<blocks, 256, 0, stream>>>(x, ws,
                                         b[0], b[1], b[2], b[3], b[4], b[5], b[6], b[7],
                                         bout, outp);
}
